// NodeTaskHeadLocal_54125177864981
// MI455X (gfx1250) — matrix operand order repaired and checked by reading, not compiled, not run on hardware
//
#include <hip/hip_runtime.h>

typedef __attribute__((ext_vector_type(16))) _Float16 v16h;
typedef __attribute__((ext_vector_type(8)))  _Float16 v8h;
typedef __attribute__((ext_vector_type(8)))  float    v8f;
typedef int v4i __attribute__((vector_size(16)));

#define NF 4
#define NLOC 2048
#define NNEI 64
#define EMBED 128
#define PAIR_DIM 64
#define NH 8
#define HD 16
#define ROWS (NF*NLOC)
#define EPS 1e-5f

#define NQK_FRAGS (17*4*32)     // 17 tiles x 4 ksteps x 32 lanes
#define NB_FRAGS  (2*32)        // bias panel: 2 ksteps x 32 lanes

#ifndef __has_builtin
#define __has_builtin(x) 0
#endif
#if __has_builtin(__builtin_amdgcn_global_load_async_to_lds_b128) && \
    __has_builtin(__builtin_amdgcn_s_wait_asynccnt)
#define USE_ASYNC_LDS 1
#else
#define USE_ASYNC_LDS 0
#endif

#if USE_ASYNC_LDS
// global_load_async_to_lds_b128: param0 = AS1 int4* (global src),
// param1 = LDS dest, imm offset, imm cpol. Tracked by ASYNCcnt.
__device__ __forceinline__ void async_copy16(const void* g, void* l) {
  __builtin_amdgcn_global_load_async_to_lds_b128(
      (__attribute__((address_space(1))) v4i*)g,
      (__attribute__((address_space(3))) v4i*)l, 0, 0);
}
#endif

// A-fragment (16x32 f16) from LDS. Lane holds row m; its 16 K-values are
// row[kb..kb+7] ++ row[kb+16..kb+23], kb = 32*kstep + ((lane&16)?8:0).
// Both halves contiguous & 16B aligned -> two ds_load_b128.
__device__ __forceinline__ v16h load_a_frag_lds(const _Float16* base, int ld, int m, int kb) {
  const _Float16* row = base + (size_t)m * ld + kb;
  v8h lo = *(const v8h*)(row);
  v8h hi = *(const v8h*)(row + 16);
  return __builtin_shufflevector(lo, hi, 0,1,2,3,4,5,6,7,8,9,10,11,12,13,14,15);
}

// ---- Kernel 0: pack all B operands into per-lane WMMA fragment layout -----
__global__ void __launch_bounds__(256)
k0_pack(const float* __restrict__ Wq, const float* __restrict__ Wk,
        const float* __restrict__ Wv, const float* __restrict__ Wf,
        const float* __restrict__ Wbias,
        _Float16* __restrict__ panelQK, _Float16* __restrict__ panelBias)
{
  int fid = blockIdx.x * 256 + threadIdx.x;
  if (fid < NQK_FRAGS) {
    int lane = fid & 31;
    int s    = (fid >> 5) & 3;
    int tile = fid >> 7;
    int n    = lane & 15;
    int kb2  = 32*s + ((lane & 16) ? 8 : 0);
    _Float16* dst = panelQK + (size_t)fid * 16;
    if (tile < 16) {
      const float* W = (tile < 8) ? Wq : Wk;
      const float* src = W + (size_t)((tile & 7)*16 + n) * EMBED + kb2;
#pragma unroll
      for (int e = 0; e < 16; ++e) dst[e] = (_Float16)src[(e < 8) ? e : e + 8];
    } else {
#pragma unroll
      for (int e = 0; e < 16; ++e) {
        float v = 0.f;
        if (n < NH) {
          int k = kb2 + ((e < 8) ? e : e + 8);
#pragma unroll
          for (int d = 0; d < HD; ++d) v += Wv[(size_t)(n*HD + d)*EMBED + k] * Wf[n*HD + d];
        }
        dst[e] = (_Float16)v;
      }
    }
  } else if (fid < NQK_FRAGS + NB_FRAGS) {
    int bid  = fid - NQK_FRAGS;
    int lane = bid & 31;
    int s    = bid >> 5;
    int n    = lane & 15;
    int kb2  = 32*s + ((lane & 16) ? 8 : 0);
    _Float16* dst = panelBias + (size_t)bid * 16;
#pragma unroll
    for (int e = 0; e < 16; ++e)
      dst[e] = (n < NH) ? (_Float16)Wbias[(size_t)n * PAIR_DIM + kb2 + ((e < 8) ? e : e + 8)] : (_Float16)0.f;
  }
}

// ---------------- Kernel 1: LN(query) -> q (scaled), k, vw via WMMA --------
__global__ void __launch_bounds__(256)
k1_ln_qkv(const float* __restrict__ query,
          const float* __restrict__ ln_g, const float* __restrict__ ln_b,
          const _Float16* __restrict__ panelQK,
          float* __restrict__ qout, float* __restrict__ kout, float* __restrict__ vwout)
{
  __shared__ __align__(16) float    s_x[16][EMBED];
  __shared__ __align__(16) _Float16 s_qn[16][EMBED];
  __shared__ float s_sum[16][16], s_sq[16][16];
  __shared__ float s_mr[16][2];

  const int tid = threadIdx.x;
  const int m0  = blockIdx.x * 16;

  // stage 16 query rows (8KB)
#if USE_ASYNC_LDS
  {
    const char* gsrc = (const char*)(query + (size_t)m0 * EMBED);
    char* ldst = (char*)&s_x[0][0];
#pragma unroll
    for (int i = 0; i < 2; ++i) {
      int e = tid + i*256;
      async_copy16(gsrc + (size_t)e*16, ldst + e*16);
    }
    __builtin_amdgcn_s_wait_asynccnt(0);
  }
#else
  {
    const float4* src = (const float4*)(query + (size_t)m0 * EMBED);
    float4* dst = (float4*)&s_x[0][0];
#pragma unroll
    for (int i = 0; i < 2; ++i) dst[tid + i*256] = src[tid + i*256];
  }
#endif
  __syncthreads();

  // LayerNorm over 128 (16 threads per row)
  {
    int row = tid >> 4, c0 = tid & 15;
    float s = 0.f, q2 = 0.f;
#pragma unroll
    for (int i = 0; i < 8; ++i) { float x = s_x[row][c0 + 16*i]; s += x; q2 += x*x; }
    s_sum[row][c0] = s; s_sq[row][c0] = q2;
  }
  __syncthreads();
  if ((tid & 15) == 0) {
    int row = tid >> 4;
    float s = 0.f, q2 = 0.f;
    for (int i = 0; i < 16; ++i) { s += s_sum[row][i]; q2 += s_sq[row][i]; }
    float mean = s * (1.f/EMBED);
    float var  = q2 * (1.f/EMBED) - mean*mean;
    s_mr[row][0] = mean; s_mr[row][1] = rsqrtf(var + EPS);
  }
  __syncthreads();
  {
    int row = tid >> 4, c0 = tid & 15;
    float mean = s_mr[row][0], r = s_mr[row][1];
#pragma unroll
    for (int i = 0; i < 8; ++i) {
      int c = c0 + 16*i;
      s_qn[row][c] = (_Float16)((s_x[row][c] - mean) * r * ln_g[c] + ln_b[c]);
    }
  }
  __syncthreads();

  const int lane   = tid & 31, wave = tid >> 5;
  const int am     = lane & 15;
  const int akoff  = (lane & 16) ? 8 : 0;
  const int bn     = lane & 15;
  const int crbase = (lane & 16) ? 8 : 0;
  const v16h* panel = (const v16h*)panelQK;

  // A fragments loaded once, reused by all tiles of this wave
  v16h a[4];
#pragma unroll
  for (int s = 0; s < 4; ++s) a[s] = load_a_frag_lds(&s_qn[0][0], EMBED, am, 32*s + akoff);

  // prefetch all B fragments of a tile, then 4 chained WMMAs
  auto run_tile = [&](int tile) -> v8f {
    v16h b0 = panel[(tile*4 + 0)*32 + lane];
    v16h b1 = panel[(tile*4 + 1)*32 + lane];
    v16h b2 = panel[(tile*4 + 2)*32 + lane];
    v16h b3 = panel[(tile*4 + 3)*32 + lane];
    v8f c = {};
    c = __builtin_amdgcn_wmma_f32_16x16x32_f16(false, a[0], false, b0, (short)0, c, false, false);
    c = __builtin_amdgcn_wmma_f32_16x16x32_f16(false, a[1], false, b1, (short)0, c, false, false);
    c = __builtin_amdgcn_wmma_f32_16x16x32_f16(false, a[2], false, b2, (short)0, c, false, false);
    c = __builtin_amdgcn_wmma_f32_16x16x32_f16(false, a[3], false, b3, (short)0, c, false, false);
    return c;
  };

  {                                 // Q tile (ntile = wave), scale HD^-0.5
    v8f c = run_tile(wave);
#pragma unroll
    for (int r = 0; r < 8; ++r)
      qout[(size_t)(m0 + crbase + r) * EMBED + wave*16 + bn] = c[r] * 0.25f;
  }
  {                                 // K tile (ntile = wave)
    v8f c = run_tile(8 + wave);
#pragma unroll
    for (int r = 0; r < 8; ++r)
      kout[(size_t)(m0 + crbase + r) * EMBED + wave*16 + bn] = c[r];
  }
  if (wave == 0) {                  // vw tile (Wforce folded into Wv)
    v8f c = run_tile(16);
    if (bn < NH) {
#pragma unroll
      for (int r = 0; r < 8; ++r)
        vwout[(size_t)(m0 + crbase + r) * NH + bn] = c[r];
    }
  }
}

// ---------------- Kernel 2: per-(f,l) attention + bias WMMA + force --------
__global__ void __launch_bounds__(256)
k2_attn(const float* __restrict__ pair, const int* __restrict__ nlist,
        const float* __restrict__ delta, const float* __restrict__ mask,
        const float* __restrict__ pn_g, const float* __restrict__ pn_b,
        const _Float16* __restrict__ panelBias, const float* __restrict__ bbias,
        const float* __restrict__ qbuf, const float* __restrict__ kbuf,
        const float* __restrict__ vwbuf, float* __restrict__ out)
{
  __shared__ __align__(16) float    s_pair[NNEI][PAIR_DIM];
  __shared__ __align__(16) _Float16 s_pn[NNEI][PAIR_DIM];
  __shared__ __align__(16) float    s_q[EMBED];
  __shared__ int   s_nl[NNEI];
  __shared__ float s_delta[NNEI][3];
  __shared__ float s_mask[NH][NNEI];
  __shared__ float s_attn[NH][NNEI];
  __shared__ float s_bias[NNEI][NH];
  __shared__ float s_vw[NNEI][NH];
  __shared__ float s_t[NNEI];
  __shared__ float s_sum[NNEI][4], s_sq[NNEI][4];
  __shared__ float s_mr[NNEI][2];

  const int tid = threadIdx.x;
  const size_t fl = blockIdx.x;
  const int f = (int)(fl >> 11), l = (int)(fl & 2047);

  // stage pair tile (16KB) via the async-LDS engine; mask rows coalesced
#if USE_ASYNC_LDS
  {
    const char* gsrc = (const char*)(pair + fl * (size_t)(NNEI*PAIR_DIM));
    char* ldst = (char*)&s_pair[0][0];
#pragma unroll
    for (int i = 0; i < 4; ++i) {
      int e = tid + i*256;
      async_copy16(gsrc + (size_t)e*16, ldst + e*16);
    }
  }
#else
  {
    const float4* psrc = (const float4*)(pair + fl * (size_t)(NNEI*PAIR_DIM));
    float4* pdst = (float4*)&s_pair[0][0];
#pragma unroll
    for (int i = 0; i < 4; ++i) pdst[tid + i*256] = psrc[tid + i*256];
  }
#endif
  if (tid < EMBED)  s_q[tid]  = qbuf[fl * EMBED + tid];
  if (tid < NNEI)   s_nl[tid] = nlist[fl * NNEI + tid];
  if (tid < 3*NNEI) ((float*)&s_delta[0][0])[tid] = delta[fl * (size_t)(3*NNEI) + tid];
#pragma unroll
  for (int i = 0; i < 2; ++i) {           // mask[f, :, l, :] -> LDS, coalesced
    int e = tid + i*256; int h = e >> 6, n = e & 63;
    s_mask[h][n] = mask[(((size_t)f*NH + h)*NLOC + l)*NNEI + n];
  }
#if USE_ASYNC_LDS
  __builtin_amdgcn_s_wait_asynccnt(0);
#endif
  __syncthreads();

  // LayerNorm of pair rows (4 threads per row of 64)
  {
    int row = tid >> 2, q4 = tid & 3;
    float s = 0.f, q2 = 0.f;
#pragma unroll
    for (int i = 0; i < 16; ++i) { float x = s_pair[row][q4*16 + i]; s += x; q2 += x*x; }
    s_sum[row][q4] = s; s_sq[row][q4] = q2;
  }
  __syncthreads();
  if ((tid & 3) == 0) {
    int row = tid >> 2;
    float s = 0.f, q2 = 0.f;
    for (int i = 0; i < 4; ++i) { s += s_sum[row][i]; q2 += s_sq[row][i]; }
    float mean = s * (1.f/PAIR_DIM);
    float var  = q2 * (1.f/PAIR_DIM) - mean*mean;
    s_mr[row][0] = mean; s_mr[row][1] = rsqrtf(var + EPS);
  }
  __syncthreads();
  {
    int row = tid >> 2, q4 = tid & 3;
    float mean = s_mr[row][0], r = s_mr[row][1];
#pragma unroll
    for (int i = 0; i < 16; ++i) {
      int c = q4*16 + i;
      s_pn[row][c] = (_Float16)((s_pair[row][c] - mean) * r * pn_g[c] + pn_b[c]);
    }
  }
  __syncthreads();

  // waves 0-3: bias = pn @ Wbias^T (WMMA, N padded 8->16); waves 4-7: gather vw
  const int lane = tid & 31, wave = tid >> 5;
  if (wave < 4) {
    const int am    = wave*16 + (lane & 15);
    const int akoff = (lane & 16) ? 8 : 0;
    const int bn    = lane & 15;
    const v16h* panel = (const v16h*)panelBias;
    v16h b0 = panel[lane];
    v16h b1 = panel[32 + lane];
    v16h a0 = load_a_frag_lds(&s_pn[0][0], PAIR_DIM, am, akoff);
    v16h a1 = load_a_frag_lds(&s_pn[0][0], PAIR_DIM, am, 32 + akoff);
    v8f c = {};
    c = __builtin_amdgcn_wmma_f32_16x16x32_f16(false, a0, false, b0, (short)0, c, false, false);
    c = __builtin_amdgcn_wmma_f32_16x16x32_f16(false, a1, false, b1, (short)0, c, false, false);
    if (bn < NH) {
      const float bb = bbias[bn];
      const int rb = (lane & 16) ? 8 : 0;
#pragma unroll
      for (int r = 0; r < 8; ++r) s_bias[wave*16 + rb + r][bn] = c[r] + bb;
    }
  } else {
    int t2 = tid - 128;
#pragma unroll
    for (int i = 0; i < 4; ++i) {
      int idx = t2*4 + i; int n = idx >> 3, h = idx & 7;
      s_vw[n][h] = vwbuf[(size_t)s_nl[n] * NH + h];
    }
  }
  __syncthreads();

  // attention logits: q.k gathered from L2-resident k table (float4 dots)
#pragma unroll
  for (int u = 0; u < 2; ++u) {
    int idx = tid*2 + u; int n = idx >> 3, h = idx & 7;
    const float4* k4 = (const float4*)(kbuf + (size_t)s_nl[n] * EMBED + h*HD);
    const float4* q4 = (const float4*)(s_q + h*HD);
    float s = 0.f;
#pragma unroll
    for (int j = 0; j < 4; ++j) {
      float4 kk = k4[j], qq = q4[j];
      s += qq.x*kk.x + qq.y*kk.y + qq.z*kk.z + qq.w*kk.w;
    }
    s += s_bias[n][h] + s_mask[h][n];
    s_attn[h][n] = s;
  }
  __syncthreads();

  // softmax: wave w handles head w (64 values, 2 per lane), wave32 shuffles
  {
    int h = wave, n0 = lane, n1 = lane + 32;
    float a0 = s_attn[h][n0], a1 = s_attn[h][n1];
    float m = fmaxf(a0, a1);
#pragma unroll
    for (int off = 16; off > 0; off >>= 1) m = fmaxf(m, __shfl_xor(m, off, 32));
    float e0 = expf(a0 - m), e1 = expf(a1 - m);
    float sum = e0 + e1;
#pragma unroll
    for (int off = 16; off > 0; off >>= 1) sum += __shfl_xor(sum, off, 32);
    float inv = 1.f / sum;
    s_attn[h][n0] = e0 * inv;
    s_attn[h][n1] = e1 * inv;
  }
  __syncthreads();

  // t[n] = sum_h probs[h,n] * vw[n,h]   (Wforce already folded into vw)
  if (tid < NNEI) {
    float t = 0.f;
#pragma unroll
    for (int h = 0; h < NH; ++h) t += s_attn[h][tid] * s_vw[tid][h];
    s_t[tid] = t;
  }
  __syncthreads();

  // force[c] = sum_n delta[n,c] * t[n]
  if (tid < 3) {
    float s = 0.f;
    for (int n = 0; n < NNEI; ++n) s += s_delta[n][tid] * s_t[n];
    out[fl*3 + tid] = s;
  }
}

extern "C" void kernel_launch(void* const* d_in, const int* in_sizes, int n_in,
                              void* d_out, int out_size, void* d_ws, size_t ws_size,
                              hipStream_t stream) {
  const float* query = (const float*)d_in[0];
  const float* pair  = (const float*)d_in[1];
  const int*   nlist = (const int*)d_in[2];
  const float* delta = (const float*)d_in[3];
  const float* mask  = (const float*)d_in[4];
  const float* ln_g  = (const float*)d_in[5];
  const float* ln_b  = (const float*)d_in[6];
  const float* pn_g  = (const float*)d_in[7];
  const float* pn_b  = (const float*)d_in[8];
  const float* Wq    = (const float*)d_in[9];
  const float* Wk    = (const float*)d_in[10];
  const float* Wv    = (const float*)d_in[11];
  const float* Wbias = (const float*)d_in[12];
  const float* bbias = (const float*)d_in[13];
  const float* Wf    = (const float*)d_in[14];
  float* out = (float*)d_out;

  _Float16* panelQK   = (_Float16*)d_ws;                         // 34816 halfs
  _Float16* panelBias = panelQK + (size_t)NQK_FRAGS * 16;        // 1024 halfs
  float* qb  = (float*)(panelBias + (size_t)NB_FRAGS * 16);      // ROWS*EMBED
  float* kb  = qb + (size_t)ROWS*EMBED;                          // ROWS*EMBED
  float* vwb = kb + (size_t)ROWS*EMBED;                          // ROWS*NH

  k0_pack  <<<9,       256, 0, stream>>>(Wq, Wk, Wv, Wf, Wbias, panelQK, panelBias);
  k1_ln_qkv<<<ROWS/16, 256, 0, stream>>>(query, ln_g, ln_b, panelQK, qb, kb, vwb);
  k2_attn  <<<ROWS,    256, 0, stream>>>(pair, nlist, delta, mask, pn_g, pn_b,
                                         panelBias, bbias, qb, kb, vwb, out);
}
